// Decoder_52828097741368
// MI455X (gfx1250) — matrix operand order repaired and checked by reading, not compiled, not run on hardware
//
#include <hip/hip_runtime.h>
#include <math.h>

// ---------------------------------------------------------------------------
// Decoder with additive attention + GRU, CDNA5 (gfx1250, wave32, WMMA bf16).
// B=32, T=64, Tx=128, V=32000, E=H=1024.
// ---------------------------------------------------------------------------

#define B_  32
#define T_  64
#define TX_ 128
#define V_  32000
#define E_  1024
#define H_  1024

typedef __bf16 bf16;
typedef __attribute__((ext_vector_type(16))) __bf16 v16bf;
typedef __attribute__((ext_vector_type(8)))  __bf16 v8bf;
typedef __attribute__((ext_vector_type(8)))  float  v8f;

// round-to-nearest-even f32 -> bf16 (bit-compatible with __bf16 storage)
__device__ __forceinline__ bf16 f2bf(float f) {
    unsigned int u = __float_as_uint(f);
    u += 0x7FFFu + ((u >> 16) & 1u);
    unsigned short s = (unsigned short)(u >> 16);
    union { unsigned short u16; bf16 b; } cv; cv.u16 = s;
    return cv.b;
}

__device__ __forceinline__ float sigmoidf_(float x) {
    return 1.0f / (1.0f + __expf(-x));
}

// ---------------------------------------------------------------------------
// f32 -> bf16 bulk convert
// ---------------------------------------------------------------------------
__global__ void cvt_bf16_kernel(const float* __restrict__ src,
                                bf16* __restrict__ dst, int n) {
    int i = blockIdx.x * blockDim.x + threadIdx.x;
    if (i < n) dst[i] = f2bf(src[i]);
}

// ---------------------------------------------------------------------------
// x_emb gather: one block per (b,t) row, converted to bf16
// ---------------------------------------------------------------------------
__global__ void gather_emb_kernel(const int* __restrict__ x,
                                  const float* __restrict__ emb,
                                  bf16* __restrict__ xe) {
    int row = blockIdx.x;                 // b*T + t
    int tok = x[row];
    const float* src = emb + (size_t)tok * E_;
    bf16* dst = xe + (size_t)row * E_;
    for (int i = threadIdx.x; i < E_; i += blockDim.x)
        dst[i] = f2bf(src[i]);
}

// ---------------------------------------------------------------------------
// WMMA bf16 GEMM:  C[M,N] (f32) = A[M,K] (bf16) @ W[N,K]^T (bf16) + bias[N]
// Block = 256 threads = 8 waves: 2 waves on M x 4 waves on N.
// Wave tile = 16(M) x 64(N): 4 accumulators, A-fragment reused 4x.
// Software-pipelined K loop: loads for block k+1 issue before the 4 WMMAs of
// block k, so s_wait_loadcnt can stay > 0 and memory overlaps the matrix pipe.
// Requires: M % 32 == 0, N % 256 == 0, K % 32 == 0 (all call sites satisfy).
// ---------------------------------------------------------------------------
__global__ void wmma_gemm_bf16_kernel(const bf16* __restrict__ A,
                                      const bf16* __restrict__ W,
                                      const float* __restrict__ bias,
                                      float* __restrict__ C,
                                      int lda, int ldw, int ldc, int K) {
    const int lane  = threadIdx.x & 31;
    const int wave  = threadIdx.x >> 5;
    const int waveM = wave >> 2;          // 0..1
    const int waveN = wave & 3;           // 0..3
    const int half  = lane >> 4;          // 0..1
    const int r     = lane & 15;          // 0..15

    const int m0 = blockIdx.y * 32 + waveM * 16;
    const int n0 = blockIdx.x * 256 + waveN * 64;

    const bf16* Arow = A + (size_t)(m0 + r) * lda;
    const bf16* Wr[4];
    float bv[4];
#pragma unroll
    for (int j = 0; j < 4; ++j) {
        const int n = n0 + j * 16 + r;
        Wr[j] = W + (size_t)n * ldw;
        bv[j] = bias ? bias[n] : 0.0f;
    }

    v8f acc[4] = {};

    // ---- prologue: load K-block 0 fragments ----
    // A fragment: row m=r, K chunk [half*16, +16) as two 8-element halves
    // B fragment: col n, K chunk [half*16, +16), contiguous 32B
    v8bf a0c = *reinterpret_cast<const v8bf*>(Arow + half * 16);
    v8bf a1c = *reinterpret_cast<const v8bf*>(Arow + half * 16 + 8);
    v16bf bc[4];
#pragma unroll
    for (int j = 0; j < 4; ++j)
        bc[j] = *reinterpret_cast<const v16bf*>(Wr[j] + half * 16);

    for (int k0 = 32; k0 < K; k0 += 32) {
        // issue next block's loads first (unconditional, no branches)
        v8bf a0n = *reinterpret_cast<const v8bf*>(Arow + k0 + half * 16);
        v8bf a1n = *reinterpret_cast<const v8bf*>(Arow + k0 + half * 16 + 8);
        v16bf bn[4];
#pragma unroll
        for (int j = 0; j < 4; ++j)
            bn[j] = *reinterpret_cast<const v16bf*>(Wr[j] + k0 + half * 16);

        // speculative prefetch one block ahead on the streamed A row
        __builtin_prefetch(Arow + k0 + 32, 0, 3);

        // compute on current block
        v16bf af;
#pragma unroll
        for (int i = 0; i < 8; ++i) { af[i] = a0c[i]; af[i + 8] = a1c[i]; }
#pragma unroll
        for (int j = 0; j < 4; ++j)
            acc[j] = __builtin_amdgcn_wmma_f32_16x16x32_bf16(
                false, af, false, bc[j], (short)0, acc[j], false, false);

        // rotate buffers
        a0c = a0n; a1c = a1n;
#pragma unroll
        for (int j = 0; j < 4; ++j) bc[j] = bn[j];
    }

    // ---- epilogue compute on last block ----
    {
        v16bf af;
#pragma unroll
        for (int i = 0; i < 8; ++i) { af[i] = a0c[i]; af[i + 8] = a1c[i]; }
#pragma unroll
        for (int j = 0; j < 4; ++j)
            acc[j] = __builtin_amdgcn_wmma_f32_16x16x32_bf16(
                false, af, false, bc[j], (short)0, acc[j], false, false);
    }

#pragma unroll
    for (int j = 0; j < 4; ++j) {
#pragma unroll
        for (int vg = 0; vg < 8; ++vg) {
            const int m = m0 + vg + half * 8;       // C layout: VGPR vg -> row
            const int n = n0 + j * 16 + r;
            C[(size_t)m * ldc + n] = acc[j][vg] + bv[j];
        }
    }
}

// ---------------------------------------------------------------------------
// s0 = henc[:,0,H:] @ initW  (tiny one-shot [32,1024]x[1024,1024]) -> bf16
// ---------------------------------------------------------------------------
__global__ void s0_kernel(const float* __restrict__ henc,
                          const float* __restrict__ initW,
                          bf16* __restrict__ s_bf) {
    int idx = blockIdx.x * blockDim.x + threadIdx.x;   // b*H + j
    int b = idx >> 10, j = idx & (H_ - 1);
    const float* hrow = henc + (size_t)b * (TX_ * 2 * H_) + H_;
    float acc = 0.0f;
    for (int k = 0; k < H_; ++k)
        acc = fmaf(hrow[k], initW[(size_t)k * H_ + j], acc);
    s_bf[idx] = f2bf(acc);
}

// ---------------------------------------------------------------------------
// Attention for one step: e = tanh(q + Ua_keys) . va ; softmax over Tx ;
// ctx[b,:] = sum_tx w * henc[b,tx,:]   -> bf16 ctx for the ctx-GEMM.
// One block per batch row b, 128 threads (one per tx).
// ---------------------------------------------------------------------------
__global__ void attention_kernel(const float* __restrict__ q,
                                 const float* __restrict__ ua_keys,
                                 const float* __restrict__ va,
                                 const float* __restrict__ henc,
                                 bf16* __restrict__ ctx_bf) {
    const int b  = blockIdx.x;
    const int tx = threadIdx.x;           // 0..127

    __shared__ float red[TX_];
    __shared__ float wsh[TX_];

    const float* qrow = q + (size_t)b * H_;
    const float* urow = ua_keys + ((size_t)b * TX_ + tx) * H_;
    float e = 0.0f;
    for (int h = 0; h < H_; ++h)
        e = fmaf(tanhf(qrow[h] + urow[h]), va[h], e);

    // softmax over the 128 tx values
    red[tx] = e; __syncthreads();
    for (int s = TX_ / 2; s > 0; s >>= 1) {
        if (tx < s) red[tx] = fmaxf(red[tx], red[tx + s]);
        __syncthreads();
    }
    float mx = red[0]; __syncthreads();
    float ex = __expf(e - mx);
    red[tx] = ex; __syncthreads();
    for (int s = TX_ / 2; s > 0; s >>= 1) {
        if (tx < s) red[tx] += red[tx + s];
        __syncthreads();
    }
    float inv = 1.0f / red[0];
    wsh[tx] = ex * inv; __syncthreads();

    // ctx: 2048 outputs across 128 threads (16 each), reduce over 128 tx
    const float* hb = henc + (size_t)b * TX_ * 2 * H_;
    for (int k = tx; k < 2 * H_; k += TX_) {
        float acc = 0.0f;
        for (int t = 0; t < TX_; ++t)
            acc = fmaf(wsh[t], hb[(size_t)t * 2 * H_ + k], acc);
        ctx_bf[(size_t)b * 2 * H_ + k] = f2bf(acc);
    }
}

// ---------------------------------------------------------------------------
// GRU epilogue for step t: gi = gi_x[:,t,:] (has b_ih) + gi_ctx ; b_hh split;
// h = (1-z)*n  -> hd_bf16[:,t,:] (for output GEMM) and s_bf16 (next query).
// ---------------------------------------------------------------------------
__global__ void gru_finish_kernel(const float* __restrict__ gi_x,
                                  const float* __restrict__ gi_ctx,
                                  const float* __restrict__ b_hh,
                                  bf16* __restrict__ hd_bf,
                                  bf16* __restrict__ s_bf,
                                  int t) {
    int idx = blockIdx.x * blockDim.x + threadIdx.x;   // b*H + j
    int b = idx >> 10, j = idx & (H_ - 1);
    size_t gx = ((size_t)(b * T_ + t)) * (3 * H_);
    size_t gc = (size_t)b * (3 * H_);
    float gr = gi_x[gx + j]            + gi_ctx[gc + j];
    float gz = gi_x[gx + H_ + j]       + gi_ctx[gc + H_ + j];
    float gn = gi_x[gx + 2 * H_ + j]   + gi_ctx[gc + 2 * H_ + j];
    float rg = sigmoidf_(gr + b_hh[j]);
    float zg = sigmoidf_(gz + b_hh[H_ + j]);
    float ng = tanhf(gn + rg * b_hh[2 * H_ + j]);
    float h  = (1.0f - zg) * ng;
    bf16 hb = f2bf(h);
    hd_bf[((size_t)(b * T_ + t)) * H_ + j] = hb;
    s_bf[idx] = hb;
}

// ---------------------------------------------------------------------------
// Host-side orchestration
// ---------------------------------------------------------------------------
extern "C" void kernel_launch(void* const* d_in, const int* in_sizes, int n_in,
                              void* d_out, int out_size, void* d_ws, size_t ws_size,
                              hipStream_t stream) {
    (void)in_sizes; (void)n_in; (void)out_size; (void)ws_size;

    const int*   x      = (const int*)  d_in[0];
    const float* henc   = (const float*)d_in[1];
    const float* emb    = (const float*)d_in[2];
    const float* Wa_w   = (const float*)d_in[3];
    const float* Wa_b   = (const float*)d_in[4];
    const float* Ua_w   = (const float*)d_in[5];
    const float* Ua_b   = (const float*)d_in[6];
    const float* Va_w   = (const float*)d_in[7];
    const float* W_ih   = (const float*)d_in[8];
    const float* b_ih   = (const float*)d_in[9];
    /* W_hh (d_in[10]) is dead: reference uses h_prev = 0 */
    const float* b_hh   = (const float*)d_in[11];
    const float* out_w  = (const float*)d_in[12];
    const float* out_b  = (const float*)d_in[13];
    const float* initW  = (const float*)d_in[14];
    float* logits = (float*)d_out;

    // --- workspace carve-up (256B aligned) ---
    char* base = (char*)d_ws;
    size_t off = 0;
    auto carve = [&](size_t bytes) -> char* {
        char* p = base + off;
        off = (off + bytes + 255) & ~(size_t)255;
        return p;
    };
    bf16*  henc_bf = (bf16*) carve((size_t)B_ * TX_ * 2 * H_ * 2);
    bf16*  WaW_bf  = (bf16*) carve((size_t)H_ * H_ * 2);
    bf16*  UaW_bf  = (bf16*) carve((size_t)H_ * 2 * H_ * 2);
    bf16*  Wih_bf  = (bf16*) carve((size_t)3 * H_ * 3 * H_ * 2);
    bf16*  outw_bf = (bf16*) carve((size_t)V_ * H_ * 2);
    bf16*  xe_bf   = (bf16*) carve((size_t)B_ * T_ * E_ * 2);
    float* uak_f   = (float*)carve((size_t)B_ * TX_ * H_ * 4);
    float* gix_f   = (float*)carve((size_t)B_ * T_ * 3 * H_ * 4);
    float* q_f     = (float*)carve((size_t)B_ * H_ * 4);
    bf16*  ctx_bf  = (bf16*) carve((size_t)B_ * 2 * H_ * 2);
    float* gictx_f = (float*)carve((size_t)B_ * 3 * H_ * 4);
    bf16*  s_bf    = (bf16*) carve((size_t)B_ * H_ * 2);
    bf16*  hd_bf   = (bf16*) carve((size_t)B_ * T_ * H_ * 2);

    auto cvt = [&](const float* s, bf16* d, int n) {
        cvt_bf16_kernel<<<(n + 255) / 256, 256, 0, stream>>>(s, d, n);
    };

    // --- one-shot: bf16 weight/activation copies (stay resident in 192MB L2)
    cvt(henc,  henc_bf, B_ * TX_ * 2 * H_);
    cvt(Wa_w,  WaW_bf,  H_ * H_);
    cvt(Ua_w,  UaW_bf,  H_ * 2 * H_);
    cvt(W_ih,  Wih_bf,  3 * H_ * 3 * H_);
    cvt(out_w, outw_bf, V_ * H_);
    gather_emb_kernel<<<B_ * T_, 256, 0, stream>>>(x, emb, xe_bf);

    // --- Ua_keys = henc @ Ua_w^T + Ua_b   [4096, 1024]
    wmma_gemm_bf16_kernel<<<dim3(H_ / 256, (B_ * TX_) / 32), 256, 0, stream>>>(
        henc_bf, UaW_bf, Ua_b, uak_f, 2 * H_, 2 * H_, H_, 2 * H_);

    // --- gi_x = x_emb @ W_ih[:, :E]^T + b_ih   [2048, 3072]  (hoisted from scan)
    wmma_gemm_bf16_kernel<<<dim3((3 * H_) / 256, (B_ * T_) / 32), 256, 0, stream>>>(
        xe_bf, Wih_bf, b_ih, gix_f, E_, 3 * H_, 3 * H_, E_);

    // --- s0 = henc[:,0,H:] @ initW  -> bf16 query state
    s0_kernel<<<(B_ * H_) / 256, 256, 0, stream>>>(henc, initW, s_bf);

    // --- sequential decode: 64 steps of {q-GEMM, attention, ctx-GEMM, GRU}
    for (int t = 0; t < T_; ++t) {
        // q = s_prev @ Wa_w^T + Wa_b   [32, 1024]
        wmma_gemm_bf16_kernel<<<dim3(H_ / 256, 1), 256, 0, stream>>>(
            s_bf, WaW_bf, Wa_b, q_f, H_, H_, H_, H_);
        // attention scores + softmax + context
        attention_kernel<<<B_, TX_, 0, stream>>>(q_f, uak_f, Va_w, henc, ctx_bf);
        // gi_ctx = ctx @ W_ih[:, E:]^T   [32, 3072]  (K offset E into W_ih rows)
        wmma_gemm_bf16_kernel<<<dim3((3 * H_) / 256, 1), 256, 0, stream>>>(
            ctx_bf, Wih_bf + E_, nullptr, gictx_f, 2 * H_, 3 * H_, 3 * H_, 2 * H_);
        // GRU gates -> h_t
        gru_finish_kernel<<<(B_ * H_) / 256, 256, 0, stream>>>(
            gix_f, gictx_f, b_hh, hd_bf, s_bf, t);
    }

    // --- logits = hd @ out_w^T + out_b   [2048, 32000]  (dominant GEMM, 134 GF)
    wmma_gemm_bf16_kernel<<<dim3(V_ / 256, (B_ * T_) / 32), 256, 0, stream>>>(
        hd_bf, outw_bf, out_b, logits, H_, H_, V_, H_);
}
